// MultiHeadAttention_50723563765938
// MI455X (gfx1250) — matrix operand order repaired and checked by reading, not compiled, not run on hardware
//
#include <hip/hip_runtime.h>
#include <stdint.h>

// Problem constants (match reference)
#define B_  2
#define S_  2048
#define D_  1024
#define H_  16
#define HD_ 64

typedef __bf16         v16bf __attribute__((ext_vector_type(16)));
typedef float          v8f   __attribute__((ext_vector_type(8)));
typedef unsigned int   v4u   __attribute__((ext_vector_type(4)));
typedef float          v4f   __attribute__((ext_vector_type(4)));
typedef unsigned short v4s   __attribute__((ext_vector_type(4)));

union Frag {
    v16bf bf;
    v4u   q[2];
};

__device__ __forceinline__ unsigned short f2bf(float f) {
    unsigned int u = __float_as_uint(f);
    u += 0x7fffu + ((u >> 16) & 1u);   // round to nearest even
    return (unsigned short)(u >> 16);
}

__device__ __forceinline__ v8f vzero8() {
    v8f z;
#pragma unroll
    for (int i = 0; i < 8; ++i) z[i] = 0.0f;
    return z;
}

__device__ __forceinline__ v8f wmma_bf16(const Frag& a, const Frag& b, v8f c) {
    // D = A(16x32 bf16) x B(32x16 bf16) + C(16x16 f32)
    return __builtin_amdgcn_wmma_f32_16x16x32_bf16(
        false, a.bf, false, b.bf, (short)0, c, false, false);
}

// LDS byte offset of a generic pointer into __shared__ (AS3 pointers are
// 0-based 32-bit LDS offsets).
__device__ __forceinline__ unsigned ldsoff(const void* p) {
    return (unsigned)(uintptr_t)(__attribute__((address_space(3))) const void*)p;
}

// 16B/lane async copy global -> LDS, tracked by ASYNCcnt (CDNA5).
__device__ __forceinline__ void async_b128(const void* gsrc, unsigned lds) {
    asm volatile("global_load_async_to_lds_b128 %0, %1, off"
                 :: "v"(lds), "v"((unsigned long long)(uintptr_t)gsrc)
                 : "memory");
}

// ---------------------------------------------------------------------------
// GEMM: Y[M,N] = A[M,K] * W[N,K]^T   (y = x @ W.T, nn.Linear convention)
// A is f32 (A_BF16=false) or bf16 bits (A_BF16=true); W always f32.
// Output: bf16 bits (OUT_F32_BIAS=false) or f32 + bias (OUT_F32_BIAS=true).
// 256 threads = 8 waves (4 over M x 2 over N); tile 128x128x32.
// Software pipelined: tile k+1 global loads issue before tile k's WMMAs.
// ---------------------------------------------------------------------------
template <bool A_BF16, bool OUT_F32_BIAS>
__global__ __launch_bounds__(256)
void gemm_xwT(const void* __restrict__ Av, const float* __restrict__ W,
              void* __restrict__ Outv, const float* __restrict__ bias,
              int M, int N, int K)
{
    constexpr int BM = 128, BN = 128, BK = 32;
    __shared__ __align__(16) unsigned short sA[BM][BK];   // 8 KB
    __shared__ __align__(16) unsigned short sB[BN][BK];   // 8 KB

    const int tid  = threadIdx.x;
    const int lane = tid & 31;
    const int w    = tid >> 5;
    const int wm   = w & 3;      // wave M index (0..3): 32 rows each
    const int wn   = w >> 2;     // wave N index (0..1): 64 cols each
    const int bm   = blockIdx.y * BM;
    const int bn   = blockIdx.x * BN;

    v8f acc[2][4];
#pragma unroll
    for (int i = 0; i < 2; ++i)
#pragma unroll
        for (int j = 0; j < 4; ++j) acc[i][j] = vzero8();

    // register staging for the pipelined global->LDS path
    v4f aF[4], wF[4];
    v4u aH[2];

    auto loadGlobal = [&](int kk) {
        if constexpr (A_BF16) {
            const unsigned short* A = (const unsigned short*)Av;
#pragma unroll
            for (int i = 0; i < 2; ++i) {
                int slot = tid + i * 256;          // 512 x (8 ushort)
                int r = slot >> 2;
                int c = (slot & 3) * 8;
                aH[i] = *(const v4u*)(A + (size_t)(bm + r) * K + kk + c);
            }
        } else {
            const float* A = (const float*)Av;
#pragma unroll
            for (int i = 0; i < 4; ++i) {
                int slot = tid + i * 256;          // 1024 x float4
                int r = slot >> 3;
                int c = (slot & 7) * 4;
                aF[i] = *(const v4f*)(A + (size_t)(bm + r) * K + kk + c);
            }
        }
#pragma unroll
        for (int i = 0; i < 4; ++i) {
            int slot = tid + i * 256;
            int r = slot >> 3;
            int c = (slot & 7) * 4;
            wF[i] = *(const v4f*)(W + (size_t)(bn + r) * K + kk + c);
        }
    };

    auto stageLDS = [&]() {
        if constexpr (A_BF16) {
#pragma unroll
            for (int i = 0; i < 2; ++i) {
                int slot = tid + i * 256;
                int r = slot >> 2;
                int c = (slot & 3) * 8;
                *(v4u*)&sA[r][c] = aH[i];
            }
        } else {
#pragma unroll
            for (int i = 0; i < 4; ++i) {
                int slot = tid + i * 256;
                int r = slot >> 3;
                int c = (slot & 7) * 4;
                v4s h;
                h[0] = f2bf(aF[i][0]); h[1] = f2bf(aF[i][1]);
                h[2] = f2bf(aF[i][2]); h[3] = f2bf(aF[i][3]);
                *(v4s*)&sA[r][c] = h;
            }
        }
#pragma unroll
        for (int i = 0; i < 4; ++i) {
            int slot = tid + i * 256;
            int r = slot >> 3;
            int c = (slot & 7) * 4;
            v4s h;
            h[0] = f2bf(wF[i][0]); h[1] = f2bf(wF[i][1]);
            h[2] = f2bf(wF[i][2]); h[3] = f2bf(wF[i][3]);
            *(v4s*)&sB[r][c] = h;
        }
    };

    loadGlobal(0);

    for (int kk = 0; kk < K; kk += BK) {
        __syncthreads();            // previous tile's fragment reads finished
        stageLDS();
        if (kk + BK < K) loadGlobal(kk + BK);   // overlap with this tile's math
        __syncthreads();

        // ---- fragments in CDNA5 WMMA VGPR layout ----
        Frag a[2], b[4];
        {
            const int row16 = lane & 15;
            const int kb = (lane < 16) ? 0 : 8;    // A: K {0..7,16..23} / {8..15,24..31}
#pragma unroll
            for (int i = 0; i < 2; ++i) {
                int r = wm * 32 + i * 16 + row16;
                a[i].q[0] = *(const v4u*)&sA[r][kb];
                a[i].q[1] = *(const v4u*)&sA[r][kb + 16];
            }
            const int k0 = (lane < 16) ? 0 : 8;    // B: K {0..7,16..23} / {8..15,24..31}
#pragma unroll
            for (int j = 0; j < 4; ++j) {
                int n = wn * 64 + j * 16 + row16;
                b[j].q[0] = *(const v4u*)&sB[n][k0];
                b[j].q[1] = *(const v4u*)&sB[n][k0 + 16];
            }
        }
#pragma unroll
        for (int i = 0; i < 2; ++i)
#pragma unroll
            for (int j = 0; j < 4; ++j)
                acc[i][j] = wmma_bf16(a[i], b[j], acc[i][j]);
    }

    // ---- epilogue: C layout = lane<16 -> rows 0-7, lane>=16 -> rows 8-15 ----
    const int n16  = lane & 15;
    const int mofs = (lane < 16) ? 0 : 8;
#pragma unroll
    for (int i = 0; i < 2; ++i) {
#pragma unroll
        for (int j = 0; j < 4; ++j) {
            int gn = bn + wn * 64 + j * 16 + n16;
            float bv = OUT_F32_BIAS ? bias[gn] : 0.0f;
#pragma unroll
            for (int v = 0; v < 8; ++v) {
                int gm = bm + wm * 32 + i * 16 + mofs + v;
                float val = acc[i][j][v] + bv;
                if (OUT_F32_BIAS)
                    ((float*)Outv)[(size_t)gm * N + gn] = val;
                else
                    ((unsigned short*)Outv)[(size_t)gm * N + gn] = f2bf(val);
            }
        }
    }
}

// ---------------------------------------------------------------------------
// Fused flash-style attention. Q/K/V: bf16 [B,S,D] (D = H*HD row-major).
// Block = 256 threads = 8 waves; block owns 128 query rows of one head,
// each wave owns 16 rows. Streams keys in 64-wide chunks.
// K chunks: async global->LDS, double buffered (ASYNCcnt pipelined).
// V chunks: register-pipelined transpose scatter into sVt, double buffered.
// ---------------------------------------------------------------------------
__global__ __launch_bounds__(256)
void attn_fused(const unsigned short* __restrict__ Q,
                const unsigned short* __restrict__ Kb,
                const unsigned short* __restrict__ Vb,
                unsigned short* __restrict__ Ctx)
{
    // scores * (1/sqrt(D)) * log2(e), softmax kept in exp2 domain
    constexpr float SCALE_LOG2E = 0.045084220027780106f; // (1/32)*log2(e)
    constexpr int   NCH = S_ / 64;                       // 32 key chunks

    __shared__ __align__(16) unsigned short sK [2][64][64];   // [buf][key][hd] 16 KB
    __shared__ __align__(16) unsigned short sVt[2][64][64];   // [buf][hd][key] 16 KB
    __shared__ __align__(16) unsigned short sP [8][16][64];   // per-wave P     16 KB

    const int tid  = threadIdx.x;
    const int lane = tid & 31;
    const int w    = tid >> 5;
    const int bh   = blockIdx.y;
    const int b    = bh >> 4;          // H_ == 16
    const int h    = bh & 15;
    const int q0   = blockIdx.x * 128 + w * 16;
    const size_t headBase = (size_t)b * S_ * D_ + (size_t)h * HD_;

    // ---- Q fragments (A layout), one per 32-wide HD step, loaded once ----
    Frag qf[2];
    {
        const int qrow = q0 + (lane & 15);
        const unsigned short* qp = Q + headBase + (size_t)qrow * D_;
        const int kb = (lane < 16) ? 0 : 8;
#pragma unroll
        for (int s2 = 0; s2 < 2; ++s2) {
            qf[s2].q[0] = *(const v4u*)(qp + s2 * 32 + kb);
            qf[s2].q[1] = *(const v4u*)(qp + s2 * 32 + kb + 16);
        }
    }

    float mrow[8], lrow[8];
    v8f acc[4];
#pragma unroll
    for (int v = 0; v < 8; ++v) { mrow[v] = -1e30f; lrow[v] = 0.0f; }
#pragma unroll
    for (int t = 0; t < 4; ++t) acc[t] = vzero8();

    // per-thread staging geometry: 2 slots of 8 ushorts over a 64x64 tile
    int slotR[2], slotC[2];
#pragma unroll
    for (int i = 0; i < 2; ++i) {
        int slot = tid + i * 256;
        slotR[i] = slot >> 3;          // chunk-local key row
        slotC[i] = (slot & 7) * 8;     // hd column base
    }

    v4u vReg[2];
    auto issueK = [&](int kc, int buf) {
#pragma unroll
        for (int i = 0; i < 2; ++i)
            async_b128(Kb + headBase + (size_t)(kc + slotR[i]) * D_ + slotC[i],
                       ldsoff(&sK[buf][slotR[i]][slotC[i]]));
    };
    auto loadV = [&](int kc) {
#pragma unroll
        for (int i = 0; i < 2; ++i)
            vReg[i] = *(const v4u*)(Vb + headBase +
                                    (size_t)(kc + slotR[i]) * D_ + slotC[i]);
    };
    auto scatterV = [&](int buf) {
#pragma unroll
        for (int i = 0; i < 2; ++i) {
            union { v4u v; unsigned short u[8]; } vv;
            vv.v = vReg[i];
#pragma unroll
            for (int j = 0; j < 8; ++j)
                sVt[buf][slotC[i] + j][slotR[i]] = vv.u[j];
        }
    };

    issueK(0, 0);
    loadV(0);

    for (int ic = 0; ic < NCH; ++ic) {
        const int buf = ic & 1;
        __syncthreads();               // all waves done with both prior buffers
        scatterV(buf);                 // V(ic) regs -> sVt[buf]
        if (ic + 1 < NCH) {
            issueK((ic + 1) * 64, buf ^ 1);   // prefetch next K chunk (async)
            loadV((ic + 1) * 64);             // prefetch next V chunk (regs)
            // async ops complete in order: <=2 outstanding => chunk ic landed
            asm volatile("s_wait_asynccnt 2" ::: "memory");
        } else {
            asm volatile("s_wait_asynccnt 0" ::: "memory");
        }
        __syncthreads();               // K async + V scatter visible block-wide

        // ---- scores: 4 tiles of 16 keys, contraction over HD=64 (2 steps) ----
        v8f sc[4];
        {
            const int keycol16 = lane & 15;
            const int k0 = (lane < 16) ? 0 : 8;
#pragma unroll
            for (int t = 0; t < 4; ++t) {
                v8f c8 = vzero8();
                const int keycol = t * 16 + keycol16;
#pragma unroll
                for (int s2 = 0; s2 < 2; ++s2) {
                    Frag kf;
                    kf.q[0] = *(const v4u*)&sK[buf][keycol][s2 * 32 + k0];
                    kf.q[1] = *(const v4u*)&sK[buf][keycol][s2 * 32 + k0 + 16];
                    c8 = wmma_bf16(qf[s2], kf, c8);
                }
                sc[t] = c8 * SCALE_LOG2E;
            }
        }

        // ---- online softmax: row stats via 16-lane-half shfl reductions ----
        float scale[8];
#pragma unroll
        for (int v = 0; v < 8; ++v) {
            float x = fmaxf(fmaxf(sc[0][v], sc[1][v]), fmaxf(sc[2][v], sc[3][v]));
#pragma unroll
            for (int off = 8; off >= 1; off >>= 1)
                x = fmaxf(x, __shfl_xor(x, off, 32));
            float mnew = fmaxf(mrow[v], x);
            scale[v] = exp2f(mrow[v] - mnew);
            mrow[v] = mnew;
        }
#pragma unroll
        for (int t2 = 0; t2 < 4; ++t2)
#pragma unroll
            for (int v = 0; v < 8; ++v) acc[t2][v] *= scale[v];

        // ---- probabilities: exp2, row-sum, stash to LDS (C -> A relayout) ----
        float rsum[8];
#pragma unroll
        for (int v = 0; v < 8; ++v) rsum[v] = 0.0f;
        {
            const int m_ = (lane < 16) ? 0 : 8;
            const int n_ = lane & 15;
#pragma unroll
            for (int t = 0; t < 4; ++t)
#pragma unroll
                for (int v = 0; v < 8; ++v) {
                    float p = exp2f(sc[t][v] - mrow[v]);
                    rsum[v] += p;
                    sP[w][m_ + v][t * 16 + n_] = f2bf(p);
                }
        }
#pragma unroll
        for (int v = 0; v < 8; ++v) {
            float x = rsum[v];
#pragma unroll
            for (int off = 8; off >= 1; off >>= 1)
                x += __shfl_xor(x, off, 32);
            lrow[v] = lrow[v] * scale[v] + x;
        }

        // DS ops from one wave complete in order; make the P stores visible
        // to this wave's fragment loads and stop compiler reordering.
        asm volatile("s_wait_dscnt 0" ::: "memory");

        // ---- PV: A = P (16x64 over 2 steps), B = V^T tiles (4 over HD) ----
        {
            const int mr = lane & 15;
            const int kb = (lane < 16) ? 0 : 8;
            const int kv0 = (lane < 16) ? 0 : 8;
#pragma unroll
            for (int s2 = 0; s2 < 2; ++s2) {
                Frag pf;
                pf.q[0] = *(const v4u*)&sP[w][mr][s2 * 32 + kb];
                pf.q[1] = *(const v4u*)&sP[w][mr][s2 * 32 + kb + 16];
#pragma unroll
                for (int t2 = 0; t2 < 4; ++t2) {
                    Frag vf;
                    const int d = t2 * 16 + mr;
                    const int k0 = s2 * 32 + kv0;
                    vf.q[0] = *(const v4u*)&sVt[buf][d][k0];
                    vf.q[1] = *(const v4u*)&sVt[buf][d][k0 + 16];
                    acc[t2] = wmma_bf16(pf, vf, acc[t2]);
                }
            }
        }
    }

    // ---- normalize and write ctx (bf16) ----
#pragma unroll
    for (int v = 0; v < 8; ++v) lrow[v] = 1.0f / lrow[v];
    const int n16 = lane & 15;
    const int mo  = (lane < 16) ? 0 : 8;
#pragma unroll
    for (int t2 = 0; t2 < 4; ++t2)
#pragma unroll
        for (int v = 0; v < 8; ++v) {
            int qrow = q0 + mo + v;
            Ctx[headBase + (size_t)qrow * D_ + t2 * 16 + n16] =
                f2bf(acc[t2][v] * lrow[v]);
        }
}

// ---------------------------------------------------------------------------
extern "C" void kernel_launch(void* const* d_in, const int* in_sizes, int n_in,
                              void* d_out, int out_size, void* d_ws, size_t ws_size,
                              hipStream_t stream)
{
    (void)in_sizes; (void)n_in; (void)out_size; (void)ws_size;

    const float* query = (const float*)d_in[0];
    const float* key   = (const float*)d_in[1];
    const float* value = (const float*)d_in[2];
    const float* Wq    = (const float*)d_in[3];
    const float* Wk    = (const float*)d_in[4];
    const float* Wv    = (const float*)d_in[5];
    const float* Wo    = (const float*)d_in[6];
    const float* bo    = (const float*)d_in[7];
    float*       out   = (float*)d_out;

    const size_t tok = (size_t)B_ * S_ * D_;       // 4M elems
    unsigned short* Qb = (unsigned short*)d_ws;    // bf16 bits
    unsigned short* Kb = Qb + tok;
    unsigned short* Vb = Kb + tok;
    unsigned short* Cb = Vb + tok;                 // total 32 MB of ws

    dim3 blk(256);
    dim3 gGemm(D_ / 128, (B_ * S_) / 128);         // (8, 32)
    dim3 gAttn(S_ / 128, B_ * H_);                 // (16, 32)

    gemm_xwT<false, false><<<gGemm, blk, 0, stream>>>(query, Wq, Qb, nullptr,
                                                      B_ * S_, D_, D_);
    gemm_xwT<false, false><<<gGemm, blk, 0, stream>>>(key,   Wk, Kb, nullptr,
                                                      B_ * S_, D_, D_);
    gemm_xwT<false, false><<<gGemm, blk, 0, stream>>>(value, Wv, Vb, nullptr,
                                                      B_ * S_, D_, D_);
    attn_fused<<<gAttn, blk, 0, stream>>>(Qb, Kb, Vb, Cb);
    gemm_xwT<true, true><<<gGemm, blk, 0, stream>>>(Cb, Wo, out, bo,
                                                    B_ * S_, D_, D_);
}
